// GNNAutoEncoder_35880156790969
// MI455X (gfx1250) — compile-verified, matrix operand order checked by reading, not run on hardware
//
#include <hip/hip_runtime.h>

#define TPB 256

typedef __bf16 bf16_t;
typedef __attribute__((ext_vector_type(16))) __bf16 v16bf;
typedef __attribute__((ext_vector_type(8)))  float  v8f;

__device__ __forceinline__ void atomic_add_f32(float* p, float v) {
  __hip_atomic_fetch_add(p, v, __ATOMIC_RELAXED, __HIP_MEMORY_SCOPE_AGENT);
}

// deg[i] = 1 (self loop)
__global__ __launch_bounds__(TPB) void k_init_deg(float* __restrict__ deg, int n) {
  int i = blockIdx.x * TPB + threadIdx.x;
  if (i < n) deg[i] = 1.0f;
}

// deg[dst[e]] += 1
__global__ __launch_bounds__(TPB) void k_accum_deg(const int* __restrict__ dst,
                                                   float* __restrict__ deg, int e) {
  int i = blockIdx.x * TPB + threadIdx.x;
  if (i < e) atomic_add_f32(&deg[dst[i]], 1.0f);
}

// in-place deg -> rsqrt(deg)
__global__ __launch_bounds__(TPB) void k_dinv(float* __restrict__ deg, int n) {
  int i = blockIdx.x * TPB + threadIdx.x;
  if (i < n) {
    float d = deg[i];
    deg[i] = (d > 0.0f) ? rsqrtf(d) : 0.0f;
  }
}

// norm[e] = dinv[src]*dinv[dst]; tail E..E+N are the self loops
__global__ __launch_bounds__(TPB) void k_norm(const int* __restrict__ src,
                                              const int* __restrict__ dst,
                                              const float* __restrict__ dinv,
                                              float* __restrict__ norm, int e, int n) {
  int i = blockIdx.x * TPB + threadIdx.x;
  if (i < e) {
    norm[i] = dinv[src[i]] * dinv[dst[i]];
  } else if (i < e + n) {
    float v = dinv[i - e];
    norm[i] = v * v;
  }
}

// H = (RELU? max(A,0):A) @ W ; A:[M,K] f32, W:[K,Hout] f32.
// W is pre-swizzled into WMMA B-fragment order in LDS (bf16), so each B
// fragment is one contiguous 32B read. One wave owns a 16-row strip and
// iterates over all column tiles, reusing the A fragment per K chunk.
// Requires M % 16 == 0 (N = 100000 = 6250*16).
template <int K, int HOUT, int RELU>
__global__ __launch_bounds__(TPB) void k_gemm_wmma(const float* __restrict__ A,
                                                   const float* __restrict__ W,
                                                   float* __restrict__ Hd, int M) {
  constexpr int KC = K >> 5;    // 32-wide K chunks
  constexpr int CT = HOUT >> 4; // 16-wide column tiles
  __shared__ __align__(32) bf16_t sW[K * HOUT];

  // Fragment-ordered fill: sW[((ct*KC + kc)*32 + lane)*16 + i] = W[k, col]
  for (int o = threadIdx.x; o < K * HOUT; o += TPB) {
    int i    = o & 15;
    int lane = (o >> 4) & 31;
    int kc   = (o >> 9) % KC;
    int ct   = o / (KC << 9);
    int half = lane >> 4;
    int col  = ct * 16 + (lane & 15);
    int k    = (kc << 5) + half * 8 + ((i < 8) ? i : (8 + i)); // runs {0..7},{16..23}
    sW[o] = (bf16_t)W[k * HOUT + col];
  }
  __syncthreads();

  int lane = threadIdx.x & 31;
  int wave = threadIdx.x >> 5;
  int row_tile = blockIdx.x * (TPB / 32) + wave;
  if (row_tile * 16 >= M) return;

  int half = lane >> 4;  // selects K sub-runs per ISA A-fragment layout
  int lrow = lane & 15;
  const float* __restrict__ arow = A + (size_t)(row_tile * 16 + lrow) * K;

  v8f acc[CT];
#pragma unroll
  for (int ct = 0; ct < CT; ++ct) acc[ct] = (v8f){0.f,0.f,0.f,0.f,0.f,0.f,0.f,0.f};

#pragma unroll
  for (int kc = 0; kc < KC; ++kc) {
    const int kb = (kc << 5) + half * 8;
    float4 a0 = *(const float4*)(arow + kb);
    float4 a1 = *(const float4*)(arow + kb + 4);
    float4 a2 = *(const float4*)(arow + kb + 16);
    float4 a3 = *(const float4*)(arow + kb + 20);
    if (RELU) {
      a0.x=fmaxf(a0.x,0.f); a0.y=fmaxf(a0.y,0.f); a0.z=fmaxf(a0.z,0.f); a0.w=fmaxf(a0.w,0.f);
      a1.x=fmaxf(a1.x,0.f); a1.y=fmaxf(a1.y,0.f); a1.z=fmaxf(a1.z,0.f); a1.w=fmaxf(a1.w,0.f);
      a2.x=fmaxf(a2.x,0.f); a2.y=fmaxf(a2.y,0.f); a2.z=fmaxf(a2.z,0.f); a2.w=fmaxf(a2.w,0.f);
      a3.x=fmaxf(a3.x,0.f); a3.y=fmaxf(a3.y,0.f); a3.z=fmaxf(a3.z,0.f); a3.w=fmaxf(a3.w,0.f);
    }
    v16bf a;
    a[0]=(bf16_t)a0.x;  a[1]=(bf16_t)a0.y;  a[2]=(bf16_t)a0.z;  a[3]=(bf16_t)a0.w;
    a[4]=(bf16_t)a1.x;  a[5]=(bf16_t)a1.y;  a[6]=(bf16_t)a1.z;  a[7]=(bf16_t)a1.w;
    a[8]=(bf16_t)a2.x;  a[9]=(bf16_t)a2.y;  a[10]=(bf16_t)a2.z; a[11]=(bf16_t)a2.w;
    a[12]=(bf16_t)a3.x; a[13]=(bf16_t)a3.y; a[14]=(bf16_t)a3.z; a[15]=(bf16_t)a3.w;

#pragma unroll
    for (int ct = 0; ct < CT; ++ct) {
      v16bf b = *reinterpret_cast<const v16bf*>(&sW[((ct * KC + kc) << 9) + (lane << 4)]);
      acc[ct] = __builtin_amdgcn_wmma_f32_16x16x32_bf16(
          false, a, false, b, (short)0, acc[ct], false, false);
    }
  }

#pragma unroll
  for (int ct = 0; ct < CT; ++ct) {
    int col = ct * 16 + lrow;
#pragma unroll
    for (int r = 0; r < 8; ++r) {  // C/D layout: VGPR r -> row r + 8*half
      Hd[(size_t)(row_tile * 16 + r + half * 8) * HOUT + col] = acc[ct][r];
    }
  }
}

// out[i, f] = bias[f]   (hout is a power of two)
__global__ __launch_bounds__(TPB) void k_bias_init(const float* __restrict__ b,
                                                   float* __restrict__ out,
                                                   long long total, int hmask) {
  long long i = (long long)blockIdx.x * TPB + threadIdx.x;
  if (i < total) out[i] = b[(int)i & hmask];
}

// out[dst[e], :] += norm[e] * h[src[e], :]   (float4 per thread, f32 global atomics)
__global__ __launch_bounds__(TPB) void k_scatter(const float* __restrict__ h,
                                                 const int* __restrict__ src,
                                                 const int* __restrict__ dst,
                                                 const float* __restrict__ norm,
                                                 float* __restrict__ out,
                                                 int e, int n, int hout, int qshift) {
  long long tid = (long long)blockIdx.x * TPB + threadIdx.x;
  long long total = (long long)(e + n) << qshift;
  if (tid >= total) return;
  int edge = (int)(tid >> qshift);
  int q = (int)tid & ((1 << qshift) - 1);
  int s, d;
  if (edge < e) { s = src[edge]; d = dst[edge]; }
  else          { s = edge - e;  d = s; }
  float w = norm[edge];
  const float4 hv = *(const float4*)(h + (size_t)s * hout + (size_t)q * 4);
  float* op = out + (size_t)d * hout + (size_t)q * 4;
  atomic_add_f32(op + 0, hv.x * w);
  atomic_add_f32(op + 1, hv.y * w);
  atomic_add_f32(op + 2, hv.z * w);
  atomic_add_f32(op + 3, hv.w * w);
}

static inline int cdiv_ll(long long a, long long b) { return (int)((a + b - 1) / b); }

extern "C" void kernel_launch(void* const* d_in, const int* in_sizes, int n_in,
                              void* d_out, int out_size, void* d_ws, size_t ws_size,
                              hipStream_t stream) {
  (void)n_in; (void)out_size; (void)ws_size;
  const int F = 128, Hh = 64, Lh = 32;

  const float* x  = (const float*)d_in[0];
  const int*   ei = (const int*)d_in[1];
  const float* W1 = (const float*)d_in[2]; const float* b1 = (const float*)d_in[3];
  const float* W2 = (const float*)d_in[4]; const float* b2 = (const float*)d_in[5];
  const float* W3 = (const float*)d_in[6]; const float* b3 = (const float*)d_in[7];
  const float* W4 = (const float*)d_in[8]; const float* b4 = (const float*)d_in[9];

  const int N = in_sizes[0] / F;
  const int E = in_sizes[1] / 2;
  const int* src = ei;
  const int* dst = ei + E;

  float* out  = (float*)d_out;
  float* xhat = out;                       // [N, F]
  float* z    = out + (size_t)N * F;       // [N, L]

  // workspace layout (floats)
  float* ws   = (float*)d_ws;
  float* dinv = ws;                        size_t o = (size_t)N;
  float* norm = ws + o;                    o += (size_t)E + N;
  float* bufA = ws + o;                    o += (size_t)N * F;   // GEMM output h
  float* bufB = ws + o;                                          // aggregated [N,64]

  // --- normalization coefficients ---
  k_init_deg<<<cdiv_ll(N, TPB), TPB, 0, stream>>>(dinv, N);
  k_accum_deg<<<cdiv_ll(E, TPB), TPB, 0, stream>>>(dst, dinv, E);
  k_dinv<<<cdiv_ll(N, TPB), TPB, 0, stream>>>(dinv, N);
  k_norm<<<cdiv_ll((long long)E + N, TPB), TPB, 0, stream>>>(src, dst, dinv, norm, E, N);

  const int gemm_blocks = cdiv_ll(N / 16, TPB / 32);

  // --- layer 1: h = x @ W1 ; z1 = agg + b1 (relu folded into layer-2 load) ---
  k_gemm_wmma<128, 64, 0><<<gemm_blocks, TPB, 0, stream>>>(x, W1, bufA, N);
  k_bias_init<<<cdiv_ll((long long)N * Hh, TPB), TPB, 0, stream>>>(b1, bufB, (long long)N * Hh, Hh - 1);
  k_scatter<<<cdiv_ll((long long)(E + N) * (Hh / 4), TPB), TPB, 0, stream>>>(
      bufA, src, dst, norm, bufB, E, N, Hh, 4);

  // --- layer 2: h = relu(z1) @ W2 ; z = agg + b2  (written straight into d_out) ---
  k_gemm_wmma<64, 32, 1><<<gemm_blocks, TPB, 0, stream>>>(bufB, W2, bufA, N);
  k_bias_init<<<cdiv_ll((long long)N * Lh, TPB), TPB, 0, stream>>>(b2, z, (long long)N * Lh, Lh - 1);
  k_scatter<<<cdiv_ll((long long)(E + N) * (Lh / 4), TPB), TPB, 0, stream>>>(
      bufA, src, dst, norm, z, E, N, Lh, 3);

  // --- layer 3: h = z @ W3 ; x2 = agg + b3 (relu folded into layer-4 load) ---
  k_gemm_wmma<32, 64, 0><<<gemm_blocks, TPB, 0, stream>>>(z, W3, bufA, N);
  k_bias_init<<<cdiv_ll((long long)N * Hh, TPB), TPB, 0, stream>>>(b3, bufB, (long long)N * Hh, Hh - 1);
  k_scatter<<<cdiv_ll((long long)(E + N) * (Hh / 4), TPB), TPB, 0, stream>>>(
      bufA, src, dst, norm, bufB, E, N, Hh, 4);

  // --- layer 4: h = relu(x2) @ W4 ; x_hat = agg + b4 ---
  k_gemm_wmma<64, 128, 1><<<gemm_blocks, TPB, 0, stream>>>(bufB, W4, bufA, N);
  k_bias_init<<<cdiv_ll((long long)N * F, TPB), TPB, 0, stream>>>(b4, xhat, (long long)N * F, F - 1);
  k_scatter<<<cdiv_ll((long long)(E + N) * (F / 4), TPB), TPB, 0, stream>>>(
      bufA, src, dst, norm, xhat, E, N, F, 5);
}
